// Decoder_20375324852593
// MI455X (gfx1250) — matrix operand order repaired and checked by reading, not compiled, not run on hardware
//
#include <hip/hip_runtime.h>
#include <math.h>

// ---------------------------------------------------------------------------
// Tacotron2-style decoder for MI455X (gfx1250).
//  * bf16 WMMA (v_wmma_f32_16x16x32_bf16) for the two per-step LSTM GEMMs
//    (B=16 == one WMMA M-tile; weights packed once to fragment-layout bf16,
//    ~36MB -> L2-resident across all 200 steps).
//  * Persistent kernel (32 blocks x 256 thr = 256 waves = 256 N-tiles of the
//    4096-wide gate matrices) with an atomic grid barrier between stages.
//  * K-loops split per A-region (pre | ctx | h) so the WMMA inner loops are
//    pure pointer-increment streams (no per-iteration SALU branch chains).
// ---------------------------------------------------------------------------

typedef __attribute__((ext_vector_type(16))) __bf16 v16bf;
typedef __attribute__((ext_vector_type(8)))  float  v8f;

#define B_      16
#define T_ENC_  256
#define T_DEC_  200
#define N_MELS_ 80
#define ENC_DIM_ 512
#define PRENET_ 256
#define ARNN_   1024
#define DRNN_   1024
#define ATT_    128
#define NF_     32
#define KS_     31

#define NGATE_  4096          // 4*1024
#define NTILE_  256           // 4096/16 N-tiles
#define KT1_    56            // K1 = 256+512+1024 = 1792 -> 56 k-frags of 32
#define KT2_    80            // K2 = 1024+512+1024 = 2560 -> 80 k-frags
#define FRAG_   512           // bf16 elements per 32x16 fragment (1KB)

__device__ __forceinline__ float sigf(float x) { return 1.0f / (1.0f + __expf(-x)); }

__device__ __forceinline__ unsigned wang_hash(unsigned s) {
  s = (s ^ 61u) ^ (s >> 16); s *= 9u; s ^= s >> 4; s *= 0x27d4eb2du; s ^= s >> 15;
  return s;
}
__device__ __forceinline__ float drop2x(float v, unsigned id, unsigned seed) {
  unsigned h = wang_hash(id * 2654435761u ^ seed);
  float r = (float)(h & 0xFFFFFFu) * (1.0f / 16777216.0f);
  return (r < 0.5f) ? v * 2.0f : 0.0f;
}

// A-fragment (16x32 bf16, ISA 7.12.2): given batch row b and k-within-tile kk,
// return lane*16+j position inside the 512-element fragment.
__device__ __forceinline__ int frag_pos(int b, int kk) {
  int hi = (kk >> 3) & 1;               // lanes 16-31 hold K 8-15 / 24-31
  int l  = b + (hi ? 16 : 0);
  int j  = (kk & 7) | ((kk & 16) ? 8 : 0);
  return l * 16 + j;
}

// Linear WMMA accumulation over one contiguous A-region.
// A frags advance by FRAG_; B frags (fixed N-tile) advance by NTILE_*FRAG_.
__device__ __forceinline__ v8f gemm_region(v8f acc, const __bf16* __restrict__ af,
                                           const __bf16* __restrict__ bf,
                                           int nkt, int lane) {
  const __bf16* ap = af + lane * 16;
  const __bf16* bp = bf + lane * 16;
#pragma unroll 4
  for (int kt = 0; kt < nkt; ++kt) {
    __builtin_prefetch(bp + 2 * (size_t)NTILE_ * FRAG_, 0, 1);
    v16bf a = *(const v16bf*)ap;
    v16bf b = *(const v16bf*)bp;
    acc = __builtin_amdgcn_wmma_f32_16x16x32_bf16(false, a, false, b,
                                                  (short)0, acc, false, false);
    ap += FRAG_;
    bp += (size_t)NTILE_ * FRAG_;
  }
  return acc;
}

// ---------------------------------------------------------------------------
// One-time prep kernels
// ---------------------------------------------------------------------------

// Pack [arnn_Wih;arnn_Whh] (K=1792) and [drnn_Wih;drnn_Whh] (K=2560) into
// bf16 B-fragments (32x16, K x N): lane l -> column n = nt*16+(l&15),
// j-th element -> k = kt*32 + (l>=16?8:0) + (j&7) + (j>=8?16:0). One thread per (frag,lane).
__global__ void pack_weights_kernel(const float* __restrict__ aWih,
                                    const float* __restrict__ aWhh,
                                    const float* __restrict__ dWih,
                                    const float* __restrict__ dWhh,
                                    __bf16* __restrict__ aFr,
                                    __bf16* __restrict__ dFr) {
  size_t id = (size_t)blockIdx.x * blockDim.x + threadIdx.x;
  const size_t total = (size_t)(KT1_ + KT2_) * NTILE_ * 32;
  if (id >= total) return;
  int l = (int)(id & 31);
  size_t fr = id >> 5;
  int isD = fr >= (size_t)KT1_ * NTILE_;
  size_t frr = isD ? fr - (size_t)KT1_ * NTILE_ : fr;
  int kt = (int)(frr / NTILE_), nt = (int)(frr % NTILE_);
  __bf16* out = (isD ? dFr : aFr) + frr * FRAG_ + l * 16;
  int kbase = kt * 32 + ((l >= 16) ? 8 : 0);
  int n = nt * 16 + (l & 15);
  for (int j = 0; j < 16; ++j) {
    int k = kbase + (j & 7) + ((j & 8) ? 16 : 0);
    float v;
    if (!isD) v = (k < 768)  ? aWih[(size_t)k * NGATE_ + n] : aWhh[(size_t)(k - 768) * NGATE_ + n];
    else      v = (k < 1536) ? dWih[(size_t)k * NGATE_ + n] : dWhh[(size_t)(k - 1536) * NGATE_ + n];
    out[j] = (__bf16)v;
  }
}

// Prenet layer 1: x1d[t][b][u] = drop2x(relu(di[b][:,t-1] @ W1)), t==0 -> 0 row
__global__ void prenet1_kernel(const float* __restrict__ di,
                               const float* __restrict__ W1,
                               float* __restrict__ x1d) {
  int id = blockIdx.x * blockDim.x + threadIdx.x;
  if (id >= T_DEC_ * B_ * PRENET_) return;
  int u = id % PRENET_, b = (id / PRENET_) % B_, t = id / (PRENET_ * B_);
  float s = 0.0f;
  if (t > 0) {
    const float* dr = di + (size_t)b * N_MELS_ * T_DEC_ + (t - 1);
    for (int p = 0; p < N_MELS_; ++p) s += dr[(size_t)p * T_DEC_] * W1[p * PRENET_ + u];
  }
  x1d[id] = drop2x(fmaxf(s, 0.0f), (unsigned)id, 0x13572468u);
}

// Prenet layer 2 + scatter straight into per-step A-fragments (bf16)
__global__ void prenet2_kernel(const float* __restrict__ x1d,
                               const float* __restrict__ W2,
                               __bf16* __restrict__ preFr) {
  int id = blockIdx.x * blockDim.x + threadIdx.x;
  if (id >= T_DEC_ * B_ * PRENET_) return;
  int u = id % PRENET_, b = (id / PRENET_) % B_, t = id / (PRENET_ * B_);
  const float* xr = x1d + (size_t)(t * B_ + b) * PRENET_;
  float s = 0.0f;
  for (int p = 0; p < PRENET_; ++p) s += xr[p] * W2[p * PRENET_ + u];
  s = drop2x(fmaxf(s, 0.0f), (unsigned)id, 0x2468ACE1u);
  int kt = u >> 5, kk = u & 31;
  preFr[((size_t)t * 8 + kt) * FRAG_ + frag_pos(b, kk)] = (__bf16)s;
}

// pm[b][p][a] = memory[b][p][:] @ Wmem[:,a]
__global__ void pm_kernel(const float* __restrict__ memory,
                          const float* __restrict__ Wmem,
                          float* __restrict__ pm) {
  int id = blockIdx.x * blockDim.x + threadIdx.x;
  if (id >= B_ * T_ENC_ * ATT_) return;
  int a = id & (ATT_ - 1);
  int p = (id >> 7) & (T_ENC_ - 1);
  int b = id >> 15;
  const float* mr = memory + ((size_t)b * T_ENC_ + p) * ENC_DIM_;
  float s = 0.0f;
  for (int e = 0; e < ENC_DIM_; ++e) s += mr[e] * Wmem[e * ATT_ + a];
  pm[id] = s;
}

// ---------------------------------------------------------------------------
// Persistent decoder: 32 blocks x 256 threads (256 waves).
// ---------------------------------------------------------------------------
__global__ void __launch_bounds__(256, 1)
decoder_persistent(const float* __restrict__ memory, const int* __restrict__ memlen,
                   const float* __restrict__ arnn_b, const float* __restrict__ Wq,
                   const float* __restrict__ v_w, const float* __restrict__ lconvW,
                   const float* __restrict__ ldenseW, const float* __restrict__ drnn_b,
                   const float* __restrict__ projW, const float* __restrict__ projb,
                   const float* __restrict__ gateW, const float* __restrict__ gateb,
                   const __bf16* __restrict__ aWfr, const __bf16* __restrict__ dWfr,
                   const __bf16* __restrict__ preFr,
                   __bf16* haFr, __bf16* ctxFr, __bf16* hdFr,
                   float* h_a, float* c_a, float* h_d, float* c_d,
                   float* ctx, float* aw, float* awc,
                   const float* __restrict__ pm, float* z_a, float* z_d,
                   float* loc, float* pq, float* ebuf,
                   float* mel_out, float* gate_out, float* align_out,
                   unsigned* syncp) {
  const int tid  = threadIdx.x;
  const int gt   = blockIdx.x * 256 + tid;
  const int wave = gt >> 5;
  const int lane = tid & 31;
  const unsigned nblocks = gridDim.x;
  unsigned epoch = 0;

  auto gbar = [&]() {
    __threadfence();
    __syncthreads();
    if (tid == 0) {
      __hip_atomic_fetch_add(syncp, 1u, __ATOMIC_RELEASE, __HIP_MEMORY_SCOPE_AGENT);
      ++epoch;
      unsigned target = epoch * nblocks;
      while (__hip_atomic_load(syncp, __ATOMIC_ACQUIRE, __HIP_MEMORY_SCOPE_AGENT) < target)
        __builtin_amdgcn_s_sleep(1);
    }
    __syncthreads();
    __threadfence();
  };

  const int nt  = wave;                  // one 16-wide N-tile per wave
  const int col = nt * 16 + (lane & 15);
  const int mhi = (lane >= 16) ? 8 : 0;

#pragma unroll 1
  for (int t = 0; t < T_DEC_; ++t) {
    // ---- GEMM1: z_a(16x4096) = [x_t, ctx, h_a](16x1792) @ arnnW
    {
      const __bf16* b0 = aWfr + (size_t)nt * FRAG_;
      v8f acc = {0.f, 0.f, 0.f, 0.f, 0.f, 0.f, 0.f, 0.f};
      acc = gemm_region(acc, preFr + (size_t)t * 8 * FRAG_, b0, 8, lane);
      acc = gemm_region(acc, ctxFr, b0 + (size_t)8 * NTILE_ * FRAG_, 16, lane);
      acc = gemm_region(acc, haFr,  b0 + (size_t)24 * NTILE_ * FRAG_, 32, lane);
#pragma unroll
      for (int r = 0; r < 8; ++r) z_a[(size_t)(r + mhi) * NGATE_ + col] = acc[r];
    }
    gbar();

    // ---- ARNN LSTM cell; scatter h into A-fragments for GEMM1(t+1)/GEMM2(t)
    for (int it = gt; it < B_ * ARNN_; it += 8192) {
      int b = it >> 10, u = it & 1023;
      const float* zr = z_a + (size_t)b * NGATE_;
      float zi = zr[u] + arnn_b[u];
      float zf = zr[1024 + u] + arnn_b[1024 + u];
      float zg = zr[2048 + u] + arnn_b[2048 + u];
      float zo = zr[3072 + u] + arnn_b[3072 + u];
      float cn = sigf(zf) * c_a[it] + sigf(zi) * tanhf(zg);
      float hn = sigf(zo) * tanhf(cn);
      c_a[it] = cn; h_a[it] = hn;
      int kt = u >> 5, kk = u & 31;
      haFr[kt * FRAG_ + frag_pos(b, kk)] = (__bf16)hn;
    }
    gbar();

    // ---- Attention part 1: location conv (aw,awc) and pq = h_a @ Wq
    for (int o = gt; o < B_ * NF_ * T_ENC_; o += 8192) {
      int p = o & 255, f = (o >> 8) & 31, b = o >> 13;
      const float* w0 = lconvW + (f * 2 + 0) * KS_;
      const float* w1 = lconvW + (f * 2 + 1) * KS_;
      const float* a0 = aw + b * T_ENC_;
      const float* a1 = awc + b * T_ENC_;
      int k0 = (p >= 15) ? 0 : 15 - p;
      int k1 = (271 - p < KS_) ? 271 - p : KS_;
      float s = 0.0f;
      for (int k = k0; k < k1; ++k) {
        int q = p + k - 15;
        s += w0[k] * a0[q] + w1[k] * a1[q];
      }
      loc[o] = s;                                   // (b*32+f)*256+p
    }
    if (gt < B_ * ATT_) {
      int a = gt & 127, b = gt >> 7;
      const float* hr = h_a + (size_t)b * ARNN_;
      float s = 0.0f;
      for (int u = 0; u < ARNN_; ++u) s += hr[u] * Wq[u * ATT_ + a];
      pq[gt] = s;
    }
    gbar();

    // ---- Attention part 2: energies (fused loc-dense) + mask
    if (gt < B_ * T_ENC_) {
      int p = gt & 255, b = gt >> 8;
      float lf[NF_];
#pragma unroll
      for (int f = 0; f < NF_; ++f) lf[f] = loc[(b * NF_ + f) * T_ENC_ + p];
      const float* pqr = pq + b * ATT_;
      const float* pmr = pm + ((size_t)b * T_ENC_ + p) * ATT_;
      float s = 0.0f;
      for (int a = 0; a < ATT_; ++a) {
        float lt = 0.0f;
#pragma unroll
        for (int f = 0; f < NF_; ++f) lt += lf[f] * ldenseW[f * ATT_ + a];
        s += tanhf(pqr[a] + lt + pmr[a]) * v_w[a];
      }
      ebuf[gt] = (p >= memlen[b]) ? -1.0e9f : s;
    }
    gbar();

    // ---- Softmax per batch row (one wave each); update aw, awc, align out
    if (wave < B_) {
      int b = wave;
      float ev[8], mx = -3.0e38f;
#pragma unroll
      for (int i = 0; i < 8; ++i) { ev[i] = ebuf[b * 256 + lane + i * 32]; mx = fmaxf(mx, ev[i]); }
      for (int o = 16; o > 0; o >>= 1) mx = fmaxf(mx, __shfl_xor(mx, o, 32));
      float sm = 0.0f;
#pragma unroll
      for (int i = 0; i < 8; ++i) { ev[i] = __expf(ev[i] - mx); sm += ev[i]; }
      for (int o = 16; o > 0; o >>= 1) sm += __shfl_xor(sm, o, 32);
      float inv = 1.0f / sm;
#pragma unroll
      for (int i = 0; i < 8; ++i) {
        int p = lane + i * 32;
        float w = ev[i] * inv;
        aw[b * 256 + p] = w;
        awc[b * 256 + p] += w;
        align_out[((size_t)b * T_DEC_ + t) * T_ENC_ + p] = w;
      }
    }
    gbar();

    // ---- Context: ctx[b][e] = sum_p w[b][p]*memory[b][p][e]; scatter frags
    {
      int e_ = gt & 511, b = gt >> 9;
      const float* wr = aw + b * T_ENC_;
      const float* mr = memory + (size_t)b * T_ENC_ * ENC_DIM_ + e_;
      float s = 0.0f;
      for (int p = 0; p < T_ENC_; ++p) s += wr[p] * mr[(size_t)p * ENC_DIM_];
      ctx[b * ENC_DIM_ + e_] = s;
      int kt = e_ >> 5, kk = e_ & 31;
      ctxFr[kt * FRAG_ + frag_pos(b, kk)] = (__bf16)s;
    }
    gbar();

    // ---- GEMM2: z_d(16x4096) = [h_a, ctx, h_d](16x2560) @ drnnW
    {
      const __bf16* b0 = dWfr + (size_t)nt * FRAG_;
      v8f acc = {0.f, 0.f, 0.f, 0.f, 0.f, 0.f, 0.f, 0.f};
      acc = gemm_region(acc, haFr,  b0, 32, lane);
      acc = gemm_region(acc, ctxFr, b0 + (size_t)32 * NTILE_ * FRAG_, 16, lane);
      acc = gemm_region(acc, hdFr,  b0 + (size_t)48 * NTILE_ * FRAG_, 32, lane);
#pragma unroll
      for (int r = 0; r < 8; ++r) z_d[(size_t)(r + mhi) * NGATE_ + col] = acc[r];
    }
    gbar();

    // ---- DRNN LSTM cell
    for (int it = gt; it < B_ * DRNN_; it += 8192) {
      int b = it >> 10, u = it & 1023;
      const float* zr = z_d + (size_t)b * NGATE_;
      float zi = zr[u] + drnn_b[u];
      float zf = zr[1024 + u] + drnn_b[1024 + u];
      float zg = zr[2048 + u] + drnn_b[2048 + u];
      float zo = zr[3072 + u] + drnn_b[3072 + u];
      float cn = sigf(zf) * c_d[it] + sigf(zi) * tanhf(zg);
      float hn = sigf(zo) * tanhf(cn);
      c_d[it] = cn; h_d[it] = hn;
      int kt = u >> 5, kk = u & 31;
      hdFr[kt * FRAG_ + frag_pos(b, kk)] = (__bf16)hn;
    }
    gbar();

    // ---- Projection: mel + gate
    if (gt < B_ * N_MELS_) {
      int m = gt % N_MELS_, b = gt / N_MELS_;
      const float* hr = h_d + (size_t)b * DRNN_;
      const float* cr = ctx + (size_t)b * ENC_DIM_;
      float s = projb[m];
      for (int k = 0; k < DRNN_; ++k)    s += hr[k] * projW[k * N_MELS_ + m];
      for (int k = 0; k < ENC_DIM_; ++k) s += cr[k] * projW[(DRNN_ + k) * N_MELS_ + m];
      mel_out[((size_t)b * N_MELS_ + m) * T_DEC_ + t] = s;
    } else if (gt < B_ * N_MELS_ + B_) {
      int b = gt - B_ * N_MELS_;
      const float* hr = h_d + (size_t)b * DRNN_;
      const float* cr = ctx + (size_t)b * ENC_DIM_;
      float s = gateb[0];
      for (int k = 0; k < DRNN_; ++k)    s += hr[k] * gateW[k];
      for (int k = 0; k < ENC_DIM_; ++k) s += cr[k] * gateW[DRNN_ + k];
      gate_out[b * T_DEC_ + t] = s;
    }
    gbar();   // end of step: state + aw/awc visible for next iteration
  }
}

// ---------------------------------------------------------------------------
extern "C" void kernel_launch(void* const* d_in, const int* in_sizes, int n_in,
                              void* d_out, int out_size, void* d_ws, size_t ws_size,
                              hipStream_t stream) {
  (void)in_sizes; (void)n_in; (void)out_size; (void)ws_size;
  const float* memory   = (const float*)d_in[0];
  const float* dec_in   = (const float*)d_in[1];
  const int*   memlen   = (const int*)d_in[2];
  const float* pW1      = (const float*)d_in[3];
  const float* pW2      = (const float*)d_in[4];
  const float* aWih     = (const float*)d_in[5];
  const float* aWhh     = (const float*)d_in[6];
  const float* arnn_b   = (const float*)d_in[7];
  const float* Wq       = (const float*)d_in[8];
  const float* Wmem     = (const float*)d_in[9];
  const float* v_w      = (const float*)d_in[10];
  const float* lconvW   = (const float*)d_in[11];
  const float* ldenseW  = (const float*)d_in[12];
  const float* dWih     = (const float*)d_in[13];
  const float* dWhh     = (const float*)d_in[14];
  const float* drnn_b   = (const float*)d_in[15];
  const float* projW    = (const float*)d_in[16];
  const float* projb    = (const float*)d_in[17];
  const float* gateW    = (const float*)d_in[18];
  const float* gateb    = (const float*)d_in[19];

  // Workspace carve-up (1KB-aligned). Zero-init region first.
  char* ws = (char*)d_ws;
  size_t off = 0;
  auto carve = [&](size_t bytes) -> char* {
    char* p = ws + off;
    off += (bytes + 1023) & ~(size_t)1023;
    return p;
  };
  unsigned* syncp = (unsigned*)carve(1024);
  __bf16* haFr  = (__bf16*)carve(32 * 1024);            // 32 frags (K=1024)
  __bf16* ctxFr = (__bf16*)carve(16 * 1024);            // 16 frags (K=512)
  __bf16* hdFr  = (__bf16*)carve(32 * 1024);
  float* h_a = (float*)carve(B_ * ARNN_ * 4);
  float* c_a = (float*)carve(B_ * ARNN_ * 4);
  float* h_d = (float*)carve(B_ * DRNN_ * 4);
  float* c_d = (float*)carve(B_ * DRNN_ * 4);
  float* ctx = (float*)carve(B_ * ENC_DIM_ * 4);
  float* aw  = (float*)carve(B_ * T_ENC_ * 4);
  float* awc = (float*)carve(B_ * T_ENC_ * 4);
  size_t zero_bytes = off;                               // everything above
  __bf16* aWfr  = (__bf16*)carve((size_t)KT1_ * NTILE_ * FRAG_ * 2);   // 14.7MB
  __bf16* dWfr  = (__bf16*)carve((size_t)KT2_ * NTILE_ * FRAG_ * 2);   // 21.0MB
  __bf16* preFr = (__bf16*)carve((size_t)T_DEC_ * 8 * FRAG_ * 2);      // 1.6MB
  float* pm   = (float*)carve((size_t)B_ * T_ENC_ * ATT_ * 4);         // 2MB
  float* z_a  = (float*)carve((size_t)B_ * NGATE_ * 4);
  float* z_d  = (float*)carve((size_t)B_ * NGATE_ * 4);
  float* x1d  = (float*)carve((size_t)T_DEC_ * B_ * PRENET_ * 4);      // 3.3MB
  float* loc  = (float*)carve((size_t)B_ * NF_ * T_ENC_ * 4);
  float* pq   = (float*)carve((size_t)B_ * ATT_ * 4);
  float* ebuf = (float*)carve((size_t)B_ * T_ENC_ * 4);

  float* out       = (float*)d_out;
  float* mel_out   = out;                                 // (16,80,200)
  float* gate_out  = out + (size_t)B_ * N_MELS_ * T_DEC_; // (16,200)
  float* align_out = gate_out + (size_t)B_ * T_DEC_;      // (16,200,256)

  // Zero sync counter + recurrent state (deterministic across graph replays)
  hipMemsetAsync(d_ws, 0, zero_bytes, stream);

  // One-time prep
  {
    size_t total = (size_t)(KT1_ + KT2_) * NTILE_ * 32;
    pack_weights_kernel<<<(unsigned)((total + 255) / 256), 256, 0, stream>>>(
        aWih, aWhh, dWih, dWhh, aWfr, dWfr);
  }
  {
    int total = T_DEC_ * B_ * PRENET_;
    prenet1_kernel<<<(total + 255) / 256, 256, 0, stream>>>(dec_in, pW1, x1d);
    prenet2_kernel<<<(total + 255) / 256, 256, 0, stream>>>(x1d, pW2, preFr);
  }
  {
    int total = B_ * T_ENC_ * ATT_;
    pm_kernel<<<(total + 255) / 256, 256, 0, stream>>>(memory, Wmem, pm);
  }

  // Persistent decoder: 32 blocks * 256 threads = 256 waves (one per N-tile)
  decoder_persistent<<<32, 256, 0, stream>>>(
      memory, memlen, arnn_b, Wq, v_w, lconvW, ldenseW, drnn_b,
      projW, projb, gateW, gateb,
      aWfr, dWfr, preFr, haFr, ctxFr, hdFr,
      h_a, c_a, h_d, c_d, ctx, aw, awc,
      pm, z_a, z_d, loc, pq, ebuf,
      mel_out, gate_out, align_out, syncp);
}
